// H2STGCN_83116207112979
// MI455X (gfx1250) — compile-verified, matrix operand order checked by reading, not run on hardware
//
#include <hip/hip_runtime.h>
#include <hip/hip_bf16.h>
#include <math.h>

// ---------------------------------------------------------------------------
// H2STGCN forward for MI455X (gfx1250, wave32, WMMA).
// Heavy op 1: per (b, t, variant)  Y[32,512] = X[32,512] @ A[512,512],
//   A = I + p * rownorm(M), M = tanh(relu(p * Net . Ne^T)) (rank-10).
//   Fused: Y = X + (X * p/D[v]) @ M, M built on the fly with
//   v_wmma_f32_16x16x32_bf16 and consumed from LDS; M never hits HBM.
//   All 4 variants (sim/comp x diag/off) in one launch via gridDim.z.
// Heavy op 2: gated st-conv = [W1|W2][64x128] @ Xcat[128xpos] on WMMA,
//   GLU gate in-register; sim+comp branches in one launch via gridDim.z.
// ---------------------------------------------------------------------------

#define NB   16          // batch
#define NT   12          // time steps
#define NN   512         // nodes
#define NH   32          // hidden
#define NE   10          // embedding dim

typedef __attribute__((ext_vector_type(16))) __bf16 v16bf;
typedef __attribute__((ext_vector_type(8)))  float  v8f;

// ---- branch-free transcendental helpers (hw TRANS ops) --------------------
__device__ __forceinline__ float fast_exp2(float x) {
#if __has_builtin(__builtin_amdgcn_exp2f)
    return __builtin_amdgcn_exp2f(x);
#else
    return exp2f(x);
#endif
}
__device__ __forceinline__ float fast_rcp(float x) {
#if __has_builtin(__builtin_amdgcn_rcpf)
    return __builtin_amdgcn_rcpf(x);
#else
    return 1.0f / x;
#endif
}
__device__ __forceinline__ float fast_tanh(float x) {      // used with x >= 0
#if __has_builtin(__builtin_amdgcn_tanhf)
    return __builtin_amdgcn_tanhf(x);                      // v_tanh_f32
#elif __has_builtin(__builtin_amdgcn_tanh_f32)
    return __builtin_amdgcn_tanh_f32(x);
#else
    float e = fast_exp2(2.885390082f * x);                 // e^(2x)
    return 1.0f - 2.0f * fast_rcp(e + 1.0f);
#endif
}
__device__ __forceinline__ float fast_sigmoid(float x) {
    return fast_rcp(1.0f + fast_exp2(-1.442695041f * x));
}

// K index inside a 16-bit WMMA A/B fragment (CDNA5 ISA layout):
// VGPR pair p in 0..3 -> K = 2p (+8 for lanes>=16), pair 4..7 -> K = 16+2(p-4).
__device__ __forceinline__ int wmma_kidx(int pair, int off, int half) {
    int base = (pair < 4) ? (pair * 2) : (16 + (pair - 4) * 2);
    return base + half * 8 + off;
}

__device__ __forceinline__ unsigned pack_bf16(float a, float b) {
    union { unsigned u; __bf16 h[2]; } p;
    p.h[0] = (__bf16)a; p.h[1] = (__bf16)b;
    return p.u;   // v_cvt_pk_bf16_f32
}

// Fragment loader from K-contiguous bf16 LDS: element (rc,k) at p[rc*ldk + k].
// K pairs are VGPR-adjacent in the fragment, so each pair is one 32-bit load.
__device__ __forceinline__ v16bf load_frag_pk(const __bf16* p, int rc, int ldk,
                                              int kmax, int half) {
    v16bf f;
#pragma unroll
    for (int pr = 0; pr < 8; ++pr) {
        int kb = wmma_kidx(pr, 0, half);
        if (kb < kmax) {
            union { unsigned u; __bf16 h[2]; } q;
            q.u = *(const unsigned*)((const char*)p + (((size_t)rc * ldk + kb) << 1));
            f[2 * pr] = q.h[0]; f[2 * pr + 1] = q.h[1];
        } else {
            f[2 * pr] = (__bf16)0.0f; f[2 * pr + 1] = (__bf16)0.0f;
        }
    }
    return f;
}

// ------------------------- LDS layout for k_graph4 -------------------------
#define LDS_XS    0                        // float [32][512]      65536 B
#define LDS_M     65536                    // bf16  [512][32] (w,v) 32768 B
#define LDS_NER   (65536 + 32768)          // bf16  [512][16]      16384 B
#define LDS_NETN  (LDS_NER + 16384)        // bf16  [512][16]      16384 B
#define LDS_DRED  (LDS_NETN + 16384)       // float [32][16]        2048 B
#define LDS_DINV  (LDS_DRED + 2048)        // float [32]             128 B
#define GRAPH_LDS_BYTES (LDS_DINV + 128)   // 133248 B (2 WG per 320KB WGP)

// ---------------------------------------------------------------------------
// k_graph4: one workgroup (16 waves, 512 thr) per (j, b, variant).
//   variant = blockIdx.z: 0 sim-diag, 1 sim-off, 2 comp-diag, 3 comp-off
//   y[b,c,w,j] = x[b,c,w,tX] + sum_v x[b,c,v,tX] * p*M[v,w]/D[v]
// diag: tL=tR=tX=j+k ; off: tL=tX=j, tR=j+k.
// ---------------------------------------------------------------------------
__global__ void k_graph4(const float* __restrict__ xsin,  // (B,32,N,T) sim
                         const float* __restrict__ xcin,  // (B,32,N,T) comp
                         const float* __restrict__ Tp,    // (B,T,N,NE)
                         const float* __restrict__ Tn,    // (B,T,N,NE)
                         const float* __restrict__ Te,    // (B,T,N,NE)
                         float* __restrict__ ysd, float* __restrict__ yso,
                         float* __restrict__ ycd, float* __restrict__ yco,
                         int k, int Tk) {
    extern __shared__ char smem[];
    float*  xs   = (float*)(smem + LDS_XS);
    __bf16* mbt  = (__bf16*)(smem + LDS_M);      // transposed: [w][v], K-contig
    __bf16* neR  = (__bf16*)(smem + LDS_NER);
    __bf16* netN = (__bf16*)(smem + LDS_NETN);
    float*  dred = (float*)(smem + LDS_DRED);
    float*  dinv = (float*)(smem + LDS_DINV);

    const int var = blockIdx.z;            // uniform per block
    const float* x   = (var < 2) ? xsin : xcin;
    const float* net = (var < 2) ? Tp : Tn;
    const float psign = (var < 2) ? 1.0f : -1.0f;
    const int diag = ((var & 1) == 0);
    float* y = (var == 0) ? ysd : (var == 1) ? yso : (var == 2) ? ycd : yco;

    const int j  = blockIdx.x;
    const int b  = blockIdx.y;
    const int tR = j + k;
    const int tL = diag ? (j + k) : j;
    const int tX = diag ? (j + k) : j;

    const int tid  = threadIdx.x;          // 0..511
    const int lane = tid & 31;
    const int wave = tid >> 5;             // 0..15
    const int half = lane >> 4;
    const int lr   = lane & 15;

    for (int i = tid; i < NH * NN; i += 512) {
        int c = i >> 9, v = i & (NN - 1);
        xs[i] = x[(((size_t)(b * NH + c) * NN + v) * NT) + tX];
    }
    {
        const float* neB  = Te  + (((size_t)b * NT + tR) * NN) * NE;
        const float* netB = net + (((size_t)b * NT + tL) * NN) * NE;
        for (int i = tid; i < NN * 16; i += 512) {
            int r = i >> 4, f = i & 15;
            neR[i]  = (__bf16)((f < NE) ? neB [r * NE + f] : 0.0f);
            netN[i] = (__bf16)((f < NE) ? netB[r * NE + f] : 0.0f);
        }
    }
    __syncthreads();

    const int wbase = wave * 32;           // this wave's 32-column strip

    // ip-WMMA B fragments of Ne^T (fixed across all v-chunks)
    v16bf bip[2];
#pragma unroll
    for (int wt = 0; wt < 2; ++wt)
        bip[wt] = load_frag_pk(neR, wbase + wt * 16 + lr, 16, 16, half);

    v8f acc[2][2];
#pragma unroll
    for (int ct = 0; ct < 2; ++ct)
#pragma unroll
        for (int wt = 0; wt < 2; ++wt) {
            v8f z = {0.f, 0.f, 0.f, 0.f, 0.f, 0.f, 0.f, 0.f};
            acc[ct][wt] = z;
        }

    for (int vc = 0; vc < NN / 32; ++vc) {
        const int v0 = vc * 32;

        // M chunk = tanh(relu(p * Net @ Ne^T)); store transposed + packed
#pragma unroll
        for (int vt = 0; vt < 2; ++vt) {
            v16bf a = load_frag_pk(netN, v0 + vt * 16 + lr, 16, 16, half);
#pragma unroll
            for (int wt = 0; wt < 2; ++wt) {
                v8f ip = {0.f, 0.f, 0.f, 0.f, 0.f, 0.f, 0.f, 0.f};
                ip = __builtin_amdgcn_wmma_f32_16x16x32_bf16(
                        false, a, false, bip[wt], (short)0, ip, false, false);
                int wc = wbase + wt * 16 + lr;
#pragma unroll
                for (int q = 0; q < 4; ++q) {
                    float m0 = fast_tanh(fmaxf(psign * ip[2 * q], 0.0f));
                    float m1 = fast_tanh(fmaxf(psign * ip[2 * q + 1], 0.0f));
                    // rows vr0 = vt*16 + 2q + 8*half (even), vr0+1
                    int u32i = wc * 16 + vt * 8 + q + 4 * half;
                    ((unsigned*)mbt)[u32i] = pack_bf16(m0, m1);
                }
            }
        }
        __syncthreads();

        // D[v] = rowsum(M[v,:]) (M >= 0); dinv = p / max(D, eps)
        {
            int r = tid >> 4, seg = tid & 15;
            float s = 0.0f;
            for (int w = seg * 32; w < seg * 32 + 32; ++w)
                s += (float)mbt[w * 32 + r];
            dred[r * 16 + seg] = s;
        }
        __syncthreads();
        if (tid < 32) {
            float s = 0.0f;
#pragma unroll
            for (int q = 0; q < 16; ++q) s += dred[tid * 16 + q];
            dinv[tid] = psign * fast_rcp(fmaxf(s, 1e-12f));
        }
        __syncthreads();

        // main GEMM: acc += (X * dinv)[16c x 32v] @ M[32v x 16w]
#pragma unroll
        for (int ct = 0; ct < 2; ++ct) {
            v16bf a;
#pragma unroll
            for (int pr = 0; pr < 8; ++pr)
#pragma unroll
                for (int off = 0; off < 2; ++off) {
                    int kk = wmma_kidx(pr, off, half);
                    float xv = xs[(ct * 16 + lr) * NN + v0 + kk] * dinv[kk];
                    a[pr * 2 + off] = (__bf16)xv;
                }
#pragma unroll
            for (int wt = 0; wt < 2; ++wt) {
                v16bf bm = load_frag_pk(mbt, wbase + wt * 16 + lr, 32, 32, half);
                acc[ct][wt] = __builtin_amdgcn_wmma_f32_16x16x32_bf16(
                        false, a, false, bm, (short)0, acc[ct][wt], false, false);
            }
        }
        __syncthreads();   // protect mbt reuse next chunk
    }

    // epilogue: y = X + acc (identity from A's eye term)
#pragma unroll
    for (int ct = 0; ct < 2; ++ct)
#pragma unroll
        for (int wt = 0; wt < 2; ++wt)
#pragma unroll
            for (int e = 0; e < 8; ++e) {
                int c = ct * 16 + e + 8 * half;
                int w = wbase + wt * 16 + lr;
                float val = xs[c * NN + w] + acc[ct][wt][e];
                y[(((size_t)(b * NH + c) * NN + w) * Tk) + j] = val;
            }
}

// ---------------------------------------------------------------------------
// Gated st-conv on WMMA, sim+comp in one launch (blockIdx.z).
// One WG (8 waves) per (position tile, b, branch):
//   H[64,128] = Wcat[64x128] @ Xcat[128x128], Xcat = [x(t); yd(j); x(j); yo(j)]
//   out[c] = (H[c]+b[c]) * sigmoid(H[c+32]+b[c+32]);  t<k -> 0.
// ---------------------------------------------------------------------------
__global__ void k_stconv2(const float* __restrict__ xsin, const float* __restrict__ xcin,
                          const float* __restrict__ ysd, const float* __restrict__ yso,
                          const float* __restrict__ ycd, const float* __restrict__ yco,
                          const float* __restrict__ W1s, const float* __restrict__ b1s,
                          const float* __restrict__ W2s, const float* __restrict__ b2s,
                          const float* __restrict__ W1c, const float* __restrict__ b1c,
                          const float* __restrict__ W2c, const float* __restrict__ b2c,
                          float* __restrict__ outs, float* __restrict__ outc,
                          int k, int Tk) {
    __shared__ __bf16 wcat[64 * 128];      // [row o][k]   16 KB
    __shared__ __bf16 xb[128 * 128];       // [col][k]     32 KB
    __shared__ float  b12[64];

    const int br = blockIdx.z;             // 0 = sim, 1 = comp (uniform)
    const float* x  = br ? xcin : xsin;
    const float* yd = br ? ycd : ysd;
    const float* yo = br ? yco : yso;
    const float* W1 = br ? W1c : W1s;
    const float* W2 = br ? W2c : W2s;
    const float* b1 = br ? b1c : b1s;
    const float* b2 = br ? b2c : b2s;
    float* out = br ? outc : outs;

    const int tid  = threadIdx.x;          // 0..255
    const int lane = tid & 31;
    const int wave = tid >> 5;             // 0..7
    const int half = lane >> 4;
    const int lr   = lane & 15;
    const int b    = blockIdx.y;
    const int pbase = blockIdx.x * 128;    // position tile (pos = n*NT + t)

    for (int i = tid; i < 64 * 128; i += 256) {
        int o = i >> 7, kk = i & 127;
        float w = (kk < 64) ? W1[o * 64 + kk] : W2[o * 64 + (kk - 64)];
        wcat[i] = (__bf16)w;
    }
    if (tid < 64) b12[tid] = b1[tid] + b2[tid];
    for (int i = tid; i < 128 * 128; i += 256) {
        int col = i >> 7, r = i & 127;
        int pos = pbase + col;
        int t = pos % NT, n = pos / NT;
        float v = 0.0f;
        if (t >= k) {
            int j = t - k;
            int c = r & 31;
            size_t base = ((size_t)(b * NH + c) * NN + n);
            if      (r < 32)  v = x [base * NT + t];
            else if (r < 64)  v = yd[base * Tk + j];
            else if (r < 96)  v = x [base * NT + j];
            else              v = yo[base * Tk + j];
        }
        xb[i] = (__bf16)v;
    }
    __syncthreads();

    const int colbase = wave * 16;
    v8f acc[4];
#pragma unroll
    for (int rt = 0; rt < 4; ++rt) {
        v8f z = {0.f, 0.f, 0.f, 0.f, 0.f, 0.f, 0.f, 0.f};
        acc[rt] = z;
    }
#pragma unroll
    for (int ks = 0; ks < 4; ++ks) {
        v16bf bfrag = load_frag_pk(xb + ks * 32, colbase + lr, 128, 32, half);
#pragma unroll
        for (int rt = 0; rt < 4; ++rt) {
            v16bf afrag = load_frag_pk(wcat + ks * 32, rt * 16 + lr, 128, 32, half);
            acc[rt] = __builtin_amdgcn_wmma_f32_16x16x32_bf16(
                    false, afrag, false, bfrag, (short)0, acc[rt], false, false);
        }
    }

    // gated epilogue (rows c in rt 0/1, gates c+32 in rt 2/3)
    int col = colbase + lr;
    int pos = pbase + col;
    int t = pos % NT, n = pos / NT;
#pragma unroll
    for (int rt = 0; rt < 2; ++rt)
#pragma unroll
        for (int e = 0; e < 8; ++e) {
            int c = rt * 16 + e + 8 * half;
            float h = acc[rt][e] + b12[c];
            float g = acc[rt + 2][e] + b12[c + 32];
            float v = (t < k) ? 0.0f : h * fast_sigmoid(g);
            out[(((size_t)(b * NH + c) * NN + n) * NT) + t] = v;
        }
}

// ---------------------------------------------------------------------------
// start conv: x[b,o,n,t] = start_W[o]*source[b,t,n,0] + start_b[o]; split 32/32
// ---------------------------------------------------------------------------
__global__ void k_start(const float* __restrict__ src, const float* __restrict__ W,
                        const float* __restrict__ bias,
                        float* __restrict__ xsim, float* __restrict__ xcomp) {
    int idx = blockIdx.x * 256 + threadIdx.x;       // B*64*N*T exact
    int t = idx % NT;
    int n = (idx / NT) % NN;
    int o = (idx / (NT * NN)) % 64;
    int b = idx / (NT * NN * 64);
    float f = src[(((size_t)b * NT + t) * NN + n) * 3 + 0];
    float v = W[o] * f + bias[o];
    if (o < NH) xsim [(((size_t)(b * NH + o)      * NN + n) * NT) + t] = v;
    else        xcomp[(((size_t)(b * NH + o - NH) * NN + n) * NT) + t] = v;
}

// temporal embedding: one thread per (b,t)
__global__ void k_temb(const float* __restrict__ src, const float* __restrict__ W1,
                       const float* __restrict__ b1, const float* __restrict__ W2,
                       const float* __restrict__ b2, float* __restrict__ temb) {
    int idx = blockIdx.x * 64 + threadIdx.x;
    if (idx >= NB * NT) return;
    int b = idx / NT, t = idx % NT;
    int day = ((int)src[(((size_t)b * NT + t) * NN + 1) * 3 + 1]) % 7;
    int tod = ((int)src[(((size_t)b * NT + t) * NN + 1) * 3 + 2]) % 288;
    if (day < 0) day += 7;
    if (tod < 0) tod += 288;
    float h[64];
#pragma unroll
    for (int d = 0; d < 64; ++d)
        h[d] = fmaxf(W1[d * 295 + day] + W1[d * 295 + 7 + tod] + b1[d], 0.0f);
    for (int g = 0; g < 8; ++g) {
        float a = b2[g];
#pragma unroll
        for (int d = 0; d < 64; ++d) a += W2[g * 64 + d] * h[d];
        temb[idx * 8 + g] = fmaxf(a, 0.0f);
    }
}

// node embedding + inter projections: one thread per (b,t,n)
__global__ void k_nodeemb(const float* __restrict__ temb,
                          const float* __restrict__ node_emb,   // (8,N,10)
                          const float* __restrict__ p_inter,    // (10,10)
                          const float* __restrict__ n_inter,
                          float* __restrict__ Te, float* __restrict__ Tp,
                          float* __restrict__ Tn) {
    int idx = blockIdx.x * 256 + threadIdx.x;       // B*T*N exact
    int bt = idx / NN, n = idx % NN;
    float tv[8];
#pragma unroll
    for (int c = 0; c < 8; ++c) tv[c] = temb[bt * 8 + c];
    float e[NE];
#pragma unroll
    for (int f = 0; f < NE; ++f) {
        float a = 0.0f;
#pragma unroll
        for (int c = 0; c < 8; ++c) a += tv[c] * node_emb[((size_t)c * NN + n) * NE + f];
        e[f] = a;
        Te[(size_t)idx * NE + f] = a;
    }
#pragma unroll
    for (int p = 0; p < NE; ++p) {
        float ap = 0.0f, an = 0.0f;
#pragma unroll
        for (int f = 0; f < NE; ++f) {
            ap += e[f] * p_inter[f * NE + p];
            an += e[f] * n_inter[f * NE + p];
        }
        Tp[(size_t)idx * NE + p] = ap;
        Tn[(size_t)idx * NE + p] = an;
    }
}

// multi-scale conv, both branches (blockIdx.y): relu(W @ [x; mp2(x); mp4(x)] + b)
__global__ void k_msc2(const float* __restrict__ xgs, const float* __restrict__ xgc,
                       const float* __restrict__ Ws, const float* __restrict__ bsx,
                       const float* __restrict__ Wc, const float* __restrict__ bcx,
                       float* __restrict__ xms, float* __restrict__ xmc) {
    const int br = blockIdx.y;
    const float* x = br ? xgc : xgs;
    const float* W = br ? Wc : Ws;
    const float* bias = br ? bcx : bsx;
    float* out = br ? xmc : xms;

    __shared__ float ws[NH * 96], bs[NH];
    int tid = threadIdx.x;
    for (int i = tid; i < NH * 96; i += 256) ws[i] = W[i];
    if (tid < NH) bs[tid] = bias[tid];
    __syncthreads();

    int idx = blockIdx.x * 256 + tid;               // B*N*T exact
    int t = idx % NT;
    int n = (idx / NT) % NN;
    int b = idx / (NT * NN);
    int t4 = t & ~3, d2 = (t & ~1) - t4, dt = t - t4;
    float xc[NH], x1[NH], x2[NH];
#pragma unroll
    for (int c = 0; c < NH; ++c) {
        size_t base = ((size_t)(b * NH + c) * NN + n) * NT + t4;
        float v0 = x[base], v1 = x[base + 1], v2 = x[base + 2], v3 = x[base + 3];
        float vv[4] = {v0, v1, v2, v3};
        xc[c] = vv[dt];
        x1[c] = fmaxf(vv[d2], vv[d2 + 1]);
        x2[c] = fmaxf(fmaxf(v0, v1), fmaxf(v2, v3));
    }
    for (int o = 0; o < NH; ++o) {
        float a = bs[o];
#pragma unroll
        for (int c = 0; c < NH; ++c)
            a += ws[o * 96 + c] * xc[c] + ws[o * 96 + NH + c] * x1[c]
               + ws[o * 96 + 64 + c] * x2[c];
        out[(((size_t)(b * NH + o) * NN + n) * NT) + t] = fmaxf(a, 0.0f);
    }
}

// ---------------------------------------------------------------------------
// mix (gated) + disentangle updates (in place into xgs/xgc) + skip (t==T-1)
// ---------------------------------------------------------------------------
__global__ void k_mix_update(const float* __restrict__ xms, const float* __restrict__ xmc,
                             float* __restrict__ xgs, float* __restrict__ xgc,
                             float* __restrict__ skip,
                             const float* __restrict__ mixW, const float* __restrict__ mixb,
                             const float* __restrict__ dsW, const float* __restrict__ dsb,
                             const float* __restrict__ dcW, const float* __restrict__ dcb,
                             const float* __restrict__ skW, const float* __restrict__ skb) {
    __shared__ float mws[64 * 64], dsws[NH * NH], dcws[NH * NH], skws[128 * NH];
    __shared__ float mbb[64], dsbb[NH], dcbb[NH];
    int tid = threadIdx.x;
    for (int i = tid; i < 64 * 64; i += 256) mws[i] = mixW[i];
    for (int i = tid; i < NH * NH; i += 256) { dsws[i] = dsW[i]; dcws[i] = dcW[i]; }
    for (int i = tid; i < 128 * NH; i += 256) skws[i] = skW[i];
    if (tid < 64) mbb[tid] = mixb[tid];
    if (tid < NH) { dsbb[tid] = dsb[tid]; dcbb[tid] = dcb[tid]; }
    __syncthreads();

    int idx = blockIdx.x * 256 + tid;               // B*N*T exact
    int t = idx % NT;
    int n = (idx / NT) % NN;
    int b = idx / (NT * NN);

    float ms[NH], mc[NH];
#pragma unroll
    for (int c = 0; c < NH; ++c) {
        size_t base = ((size_t)(b * NH + c) * NN + n) * NT + t;
        ms[c] = xms[base];
        mc[c] = xmc[base];
    }
    float xmix[NH];
    for (int c = 0; c < NH; ++c) {
        float a1 = mbb[c], a2 = mbb[NH + c];
#pragma unroll
        for (int cc = 0; cc < NH; ++cc) {
            a1 += mws[c * 64 + cc] * ms[cc] + mws[c * 64 + NH + cc] * mc[cc];
            a2 += mws[(NH + c) * 64 + cc] * ms[cc] + mws[(NH + c) * 64 + NH + cc] * mc[cc];
        }
        xmix[c] = a1 * fast_sigmoid(a2);
    }
    float diss[NH], disc[NH];
#pragma unroll
    for (int c = 0; c < NH; ++c) { diss[c] = xmix[c] - ms[c]; disc[c] = xmix[c] - mc[c]; }
    for (int c = 0; c < NH; ++c) {
        float gs = dsbb[c], gc = dcbb[c];
#pragma unroll
        for (int cc = 0; cc < NH; ++cc) {
            gs += dsws[c * NH + cc] * diss[cc];
            gc += dcws[c * NH + cc] * disc[cc];
        }
        size_t base = ((size_t)(b * NH + c) * NN + n) * NT + t;
        xgs[base] += fast_sigmoid(gs) * diss[c];
        xgc[base] += fast_sigmoid(gc) * disc[c];
    }
    if (t == NT - 1) {  // skip += sk_W @ x_mix[..., -1]  (unique thread per (b,n))
        for (int o = 0; o < 128; ++o) {
            float s = skb[o];
#pragma unroll
            for (int c = 0; c < NH; ++c) s += skws[o * NH + c] * xmix[c];
            size_t si = (size_t)(b * 128 + o) * NN + n;
            skip[si] += s;
        }
    }
}

// BN stage 1, both branches (blockIdx.z): partial sum/sumsq of (xg + res)
__global__ void k_bn_stats2(const float* __restrict__ xgs, const float* __restrict__ xgc,
                            const float* __restrict__ rs, const float* __restrict__ rc,
                            float* __restrict__ part) {
    const int br = blockIdx.z;
    const float* xg  = br ? xgc : xgs;
    const float* res = br ? rc : rs;
    float* pt = part + br * 1024;

    __shared__ float s1[256], s2[256];
    int c = blockIdx.y, tid = threadIdx.x;
    float a = 0.0f, q = 0.0f;
    const int per_ch = NB * NN * NT;                // 98304
    for (int i = blockIdx.x * 256 + tid; i < per_ch; i += 16 * 256) {
        int b = i / (NN * NT), r = i % (NN * NT);
        size_t off = (size_t)(b * NH + c) * (NN * NT) + r;
        float v = xg[off] + res[off];
        a += v; q += v * v;
    }
    s1[tid] = a; s2[tid] = q;
    __syncthreads();
    for (int s = 128; s > 0; s >>= 1) {
        if (tid < s) { s1[tid] += s1[tid + s]; s2[tid] += s2[tid + s]; }
        __syncthreads();
    }
    if (tid == 0) {
        pt[c * 16 + blockIdx.x] = s1[0];
        pt[512 + c * 16 + blockIdx.x] = s2[0];
    }
}

// BN stage 2, both branches (blockIdx.y): normalize (xg + res) -> res
__global__ void k_bn_apply2(const float* __restrict__ xgs, const float* __restrict__ xgc,
                            float* __restrict__ rs, float* __restrict__ rc,
                            const float* __restrict__ part,
                            const float* __restrict__ gs_, const float* __restrict__ bs_,
                            const float* __restrict__ gc_, const float* __restrict__ bc_) {
    const int br = blockIdx.y;
    const float* xg = br ? xgc : xgs;
    float* res = br ? rc : rs;
    const float* pt = part + br * 1024;
    const float* gamma = br ? gc_ : gs_;
    const float* beta  = br ? bc_ : bs_;

    __shared__ float mu[NH], iv[NH], gg[NH], bb[NH];
    int tid = threadIdx.x;
    if (tid < NH) {
        float s = 0.0f, q = 0.0f;
#pragma unroll
        for (int blk = 0; blk < 16; ++blk) {
            s += pt[tid * 16 + blk];
            q += pt[512 + tid * 16 + blk];
        }
        const float inv_cnt = 1.0f / (float)(NB * NN * NT);
        float m = s * inv_cnt;
        float var = q * inv_cnt - m * m;
        mu[tid] = m;
        iv[tid] = rsqrtf(var + 1e-5f);
        gg[tid] = gamma[tid];
        bb[tid] = beta[tid];
    }
    __syncthreads();
    size_t idx = (size_t)blockIdx.x * 256 + tid;    // B*32*N*T exact
    int c = (int)((idx / (NN * NT)) % NH);
    float v = xg[idx] + res[idx];
    res[idx] = gg[c] * (v - mu[c]) * iv[c] + bb[c];
}

// output head: relu(skip) -> relu(e1) -> e2 ; one thread per (b,n)
__global__ void k_head(const float* __restrict__ skip,
                       const float* __restrict__ e1W, const float* __restrict__ e1b,
                       const float* __restrict__ e2W, const float* __restrict__ e2b,
                       float* __restrict__ out) {
    int idx = blockIdx.x * 256 + threadIdx.x;       // B*N exact
    int b = idx / NN, n = idx % NN;
    float s[128];
#pragma unroll
    for (int o = 0; o < 128; ++o)
        s[o] = fmaxf(skip[(size_t)(b * 128 + o) * NN + n], 0.0f);
    float acc[NT];
#pragma unroll
    for (int t = 0; t < NT; ++t) acc[t] = e2b[t];
    for (int o = 0; o < 192; ++o) {
        float h = e1b[o];
#pragma unroll
        for (int c = 0; c < 128; ++c) h += e1W[o * 128 + c] * s[c];
        h = fmaxf(h, 0.0f);
#pragma unroll
        for (int t = 0; t < NT; ++t) acc[t] += e2W[t * 192 + o] * h;
    }
#pragma unroll
    for (int t = 0; t < NT; ++t)
        out[((size_t)b * NT + t) * NN + n] = acc[t];
}

// ---------------------------------------------------------------------------
extern "C" void kernel_launch(void* const* d_in, const int* in_sizes, int n_in,
                              void* d_out, int out_size, void* d_ws, size_t ws_size,
                              hipStream_t stream) {
    // d_in order (setup_inputs insertion order, params flattened depth-first):
    // 0 source, 1 targets, 2 start_W, 3 start_b, 4 te_W1, 5 te_b1, 6 te_W2,
    // 7 te_b2, 8 node_emb, 9 p_inter, 10 n_inter, 11 e1_W, 12 e1_b, 13 e2_W,
    // 14 e2_b, 15.. layers (24 tensors each)
    auto P = [&](int i) { return (const float*)d_in[i]; };
    const float* src = P(0);

    const size_t ACT = (size_t)NB * NH * NN * NT;   // 3,145,728 floats
    float* ws = (float*)d_ws;
    float* xsim  = ws + 0 * ACT;
    float* xcomp = ws + 1 * ACT;
    float* xgs   = ws + 2 * ACT;
    float* xgc   = ws + 3 * ACT;
    float* xms   = ws + 4 * ACT;
    float* xmc   = ws + 5 * ACT;
    float* ysd   = ws + 6 * ACT;    // graph outputs, 4 variants
    float* yso   = ws + 7 * ACT;
    float* ycd   = ws + 8 * ACT;
    float* yco   = ws + 9 * ACT;
    float* extra = ws + 10 * ACT;
    float* temb  = extra;                                   // 1536
    float* Te    = temb + 2048;                             // 983040
    float* Tp    = Te + (size_t)NB * NT * NN * NE;
    float* Tn    = Tp + (size_t)NB * NT * NN * NE;
    float* skip  = Tn + (size_t)NB * NT * NN * NE;          // 1,048,576
    float* part  = skip + (size_t)NB * 128 * NN;            // 2048 (2 branches)

    hipMemsetAsync(skip, 0, (size_t)NB * 128 * NN * sizeof(float), stream);
    hipFuncSetAttribute(reinterpret_cast<const void*>(k_graph4),
                        hipFuncAttributeMaxDynamicSharedMemorySize,
                        GRAPH_LDS_BYTES);

    // embeddings + start conv
    k_start<<<(NB * 64 * NN * NT) / 256, 256, 0, stream>>>(src, P(2), P(3), xsim, xcomp);
    k_temb<<<3, 64, 0, stream>>>(src, P(4), P(5), P(6), P(7), temb);
    k_nodeemb<<<(NB * NT * NN) / 256, 256, 0, stream>>>(temb, P(8), P(9), P(10),
                                                        Te, Tp, Tn);

    const int KS[4] = {1, 2, 4, 4};
    const int posBlocks = (NB * NN * NT) / 256;     // 384
    for (int L = 0; L < 4; ++L) {
        const int k = KS[L], Tk = NT - k, base = 15 + L * 24;

        // all 4 graph variants in one launch (sim/comp x diag/off)
        k_graph4<<<dim3(Tk, NB, 4), 512, GRAPH_LDS_BYTES, stream>>>(
            xsim, xcomp, Tp, Tn, Te, ysd, yso, ycd, yco, k, Tk);

        // gated st-conv, both branches in one launch
        k_stconv2<<<dim3(NN * NT / 128, NB, 2), 256, 0, stream>>>(
            xsim, xcomp, ysd, yso, ycd, yco,
            P(base + 0), P(base + 1), P(base + 2), P(base + 3),
            P(base + 4), P(base + 5), P(base + 6), P(base + 7),
            xgs, xgc, k, Tk);

        // multi-scale (both branches) + mix/disentangle/skip
        k_msc2<<<dim3(posBlocks, 2), 256, 0, stream>>>(
            xgs, xgc, P(base + 8), P(base + 9), P(base + 10), P(base + 11),
            xms, xmc);
        k_mix_update<<<posBlocks, 256, 0, stream>>>(xms, xmc, xgs, xgc, skip,
            P(base + 12), P(base + 13), P(base + 14), P(base + 15),
            P(base + 16), P(base + 17), P(base + 18), P(base + 19));

        // batch norms, both branches, deterministic two-stage reduction
        k_bn_stats2<<<dim3(16, NH, 2), 256, 0, stream>>>(xgs, xgc, xsim, xcomp, part);
        k_bn_apply2<<<dim3((unsigned)(ACT / 256), 2), 256, 0, stream>>>(
            xgs, xgc, xsim, xcomp, part,
            P(base + 20), P(base + 21), P(base + 22), P(base + 23));
    }

    k_head<<<(NB * NN) / 256, 256, 0, stream>>>(skip, P(11), P(12), P(13), P(14),
                                                (float*)d_out);
}
